// KimiK2_4879082848958
// MI455X (gfx1250) — matrix operand order repaired and checked by reading, not compiled, not run on hardware
//
#include <hip/hip_runtime.h>
#include <hip/hip_bf16.h>
#include <math.h>

// Model dims
#define D_    1024
#define NH_   16
#define L_    2
#define E_    16
#define KTOP_ 4
#define DFF_  1024
#define C_    10
#define B_    2
#define S_    1024
#define T_    (B_ * S_)
#define DK_   64

typedef __bf16 bf16;
typedef __attribute__((ext_vector_type(16))) __bf16 v16bf;
typedef __attribute__((ext_vector_type(8)))  __bf16 v8bf;
typedef __attribute__((ext_vector_type(8)))  float  v8f;

__device__ __forceinline__ float b2f(bf16 x) { return (float)x; }
__device__ __forceinline__ bf16  f2b(float x) { return (bf16)x; }

// ---------------- Tensor Data Mover (CDNA5 TDM) support, probe-gated -------
#if defined(__has_builtin)
#if __has_builtin(__builtin_amdgcn_tensor_load_to_lds) && \
    __has_builtin(__builtin_amdgcn_s_wait_tensorcnt)
#define HAVE_TDM 1
#endif
#endif
#ifndef HAVE_TDM
#define HAVE_TDM 0
#endif

#if HAVE_TDM
typedef unsigned int tdm_v4u __attribute__((ext_vector_type(4)));
typedef int          tdm_v8i __attribute__((ext_vector_type(8)));
typedef int          tdm_v4i __attribute__((ext_vector_type(4)));

// 2D tile load: tile_rows x tile_cols (bf16 elems, dim0 = contiguous cols),
// source rows separated by row_stride_elems. LDS gets pad_amount DWORDs of
// padding after every pad_interval DWORDs (codes per D# spec).
// Rows/cols beyond (rows, cols) read as zero (TDM OOB semantics).
__device__ __forceinline__ void tdm_load_2d(unsigned lds_off, const void* gaddr,
                                            unsigned cols, unsigned rows,
                                            unsigned tile_cols, unsigned tile_rows,
                                            unsigned long row_stride_elems,
                                            unsigned pad_int_code, unsigned pad_amt_code) {
    unsigned long ga = (unsigned long)gaddr;
    tdm_v4u g0;
    g0[0] = 1u;                                    // count=1, user mode, no gather
    g0[1] = lds_off;                               // lds_addr (bytes)
    g0[2] = (unsigned)(ga & 0xffffffffu);          // global_addr[31:0]
    g0[3] = (unsigned)((ga >> 32) & 0x01ffffffu)   // global_addr[56:32]
            | (2u << 30);                          // type = 2 ("image")
    tdm_v8i g1;
    unsigned w0 = (1u << 16)                       // data_size = 1 -> 2 bytes
                | (1u << 20)                       // pad_enable
                | (pad_int_code << 22)
                | (pad_amt_code << 25);
    g1[0] = (int)w0;
    g1[1] = (int)((cols & 0xffffu) << 16);                         // tensor_dim0[15:0]
    g1[2] = (int)(((cols >> 16) & 0xffffu) | ((rows & 0xffffu) << 16)); // dim0 hi | dim1 lo
    g1[3] = (int)(((rows >> 16) & 0xffffu) | (tile_cols << 16));   // dim1 hi | tile_dim0
    g1[4] = (int)(tile_rows & 0xffffu);                            // tile_dim1 | tile_dim2=0
    g1[5] = (int)(unsigned)(row_stride_elems & 0xffffffffu);       // dim0_stride[31:0]
    g1[6] = (int)(unsigned)((row_stride_elems >> 32) & 0xffffu);   // dim0_stride[47:32]
    g1[7] = 0;
    tdm_v4i z4;
    z4[0] = 0; z4[1] = 0; z4[2] = 0; z4[3] = 0;
    tdm_v8i z8;
    z8[0] = 0; z8[1] = 0; z8[2] = 0; z8[3] = 0;
    z8[4] = 0; z8[5] = 0; z8[6] = 0; z8[7] = 0;
    // 6-arg form (clang-23 / therock headers): g0, g1, g2, g3, extra group, cpol
    __builtin_amdgcn_tensor_load_to_lds(g0, g1, z4, z4, z8, 0);
}
#endif

// ---------------------------------------------------------------- utilities
__global__ void k_f32_to_bf16(const float* __restrict__ in, bf16* __restrict__ out, long n) {
    long i = (long)blockIdx.x * blockDim.x + threadIdx.x;
    long stride = (long)gridDim.x * blockDim.x;
    for (; i < n; i += stride) out[i] = f2b(in[i]);
}

__global__ void k_embed(const int* __restrict__ X, const float* __restrict__ emb,
                        float* __restrict__ x) {
    int t = blockIdx.x;
    long row = (long)X[t];
    for (int d = threadIdx.x; d < D_; d += blockDim.x)
        x[(long)t * D_ + d] = emb[row * D_ + d];
}

// RMSNorm over D, writes bf16 and/or f32 outputs (either may be null)
__global__ void k_rmsnorm(const float* __restrict__ x, const float* __restrict__ w,
                          bf16* __restrict__ outb, float* __restrict__ outf) {
    int t = blockIdx.x;
    const float* xr = x + (long)t * D_;
    __shared__ float red[8];
    float ss = 0.f;
    for (int d = threadIdx.x; d < D_; d += blockDim.x) { float v = xr[d]; ss += v * v; }
    for (int off = 16; off > 0; off >>= 1) ss += __shfl_down(ss, off, 32);
    int lane = threadIdx.x & 31, wid = threadIdx.x >> 5;
    if (lane == 0) red[wid] = ss;
    __syncthreads();
    if (threadIdx.x == 0) {
        float s = 0.f;
        int nw = blockDim.x >> 5;
        for (int i = 0; i < nw; ++i) s += red[i];
        red[0] = rsqrtf(s / (float)D_ + 1e-6f);
    }
    __syncthreads();
    float r = red[0];
    for (int d = threadIdx.x; d < D_; d += blockDim.x) {
        float v = xr[d] * r * w[d];
        if (outb) outb[(long)t * D_ + d] = f2b(v);
        if (outf) outf[(long)t * D_ + d] = v;
    }
}

// ---------------------------------------------------------------- WMMA GEMM
// C[M,N] (+)= A[M,K] * B[K,N], A bf16 row-major (contiguous in k),
// B bf16 generic strides. BK=true -> B contiguous along k (stride 1, col stride ldbc)
// BK=false -> B row-major (row stride ldbr, contiguous along n).
// Batched: batch index bz decomposed as (outer, inner) with divisor nInner.
#define BLK_M 128
#define BLK_N 128
#define BLK_K 32
#define LDSP  40   // padded LDS row (bf16 elems): 80B rows -> conflict-free
// TDM pad codes producing the LDSP layout: after 16 DWORDs (32 bf16) insert 4
// DWORDs (8 bf16):  interval code 3 (=16 DW), amount code 3 (=4 DW)
#define TDM_PAD_INT 3u
#define TDM_PAD_AMT 3u

enum { EPI_F32 = 0, EPI_BF16 = 1, EPI_ADD = 2, EPI_ADD_SCALED = 3 };

template <int EPI, bool BK>
__global__ __launch_bounds__(256) void k_gemm(
    const bf16* __restrict__ A, long lda, long sAo, long sAi,
    const bf16* __restrict__ Bm, long ldbr, long ldbc, long sBo, long sBi,
    void* __restrict__ Cp, long ldc, long sCo, long sCi,
    const float* __restrict__ rs, int rs_stride,
    int M, int N, int Kd, int nInner) {
    __shared__ bf16 As[BLK_M][LDSP];
    __shared__ bf16 Bs[BLK_N][LDSP];

    const int tid  = threadIdx.x;
    const int lane = tid & 31;
    const int wave = tid >> 5;
    const int wm   = wave >> 2;  // 0..1 -> 64-row strip
    const int wn   = wave & 3;   // 0..3 -> 32-col strip
    const int bz   = blockIdx.z;
    const int bo   = bz / nInner;
    const int bi   = bz - bo * nInner;
    const int M0   = blockIdx.y * BLK_M;
    const int N0   = blockIdx.x * BLK_N;

    const bf16* Ab = A + (long)bo * sAo + (long)bi * sAi;
    const bf16* Bb = Bm + (long)bo * sBo + (long)bi * sBi;

    v8f zf;
#pragma unroll
    for (int r = 0; r < 8; ++r) zf[r] = 0.f;
    v8f acc[4][2];
#pragma unroll
    for (int i = 0; i < 4; ++i)
#pragma unroll
        for (int j = 0; j < 2; ++j) acc[i][j] = zf;

    v8bf zb;
#pragma unroll
    for (int i = 0; i < 8; ++i) zb[i] = (bf16)0.0f;

    const int arow  = tid >> 1;
    const int akseg = (tid & 1) * 16;

#if HAVE_TDM
    const unsigned ldsA = (unsigned)(unsigned long)(uintptr_t)&As[0][0];
    const unsigned ldsB = (unsigned)(unsigned long)(uintptr_t)&Bs[0][0];
#endif

    for (int k0 = 0; k0 < Kd; k0 += BLK_K) {
#if HAVE_TDM
        // ---- stage A tile (128 rows x 32 k) via Tensor Data Mover
        if (wave == 0) {
            tdm_load_2d(ldsA, Ab + (long)M0 * lda + k0,
                        (unsigned)(Kd - k0), (unsigned)(M - M0),
                        BLK_K, BLK_M, (unsigned long)lda,
                        TDM_PAD_INT, TDM_PAD_AMT);
            if (BK)
                tdm_load_2d(ldsB, Bb + (long)N0 * ldbc + k0,
                            (unsigned)(Kd - k0), (unsigned)(N - N0),
                            BLK_K, BLK_N, (unsigned long)ldbc,
                            TDM_PAD_INT, TDM_PAD_AMT);
        }
#else
        // ---- stage A tile (128 x 32) manually
        {
            int m = M0 + arow;
            v8bf lo = zb, hi = zb;
            if (m < M) {
                const bf16* p = Ab + (long)m * lda + (k0 + akseg);
                lo = *(const v8bf*)(p);
                hi = *(const v8bf*)(p + 8);
            }
            *(v8bf*)(&As[arow][akseg])     = lo;
            *(v8bf*)(&As[arow][akseg + 8]) = hi;
        }
        if (BK) {
            int nn = tid >> 1;
            int kseg = (tid & 1) * 16;
            int n = N0 + nn;
            v8bf lo = zb, hi = zb;
            if (n < N) {
                const bf16* p = Bb + (long)(k0 + kseg) + (long)n * ldbc;
                lo = *(const v8bf*)(p);
                hi = *(const v8bf*)(p + 8);
            }
            *(v8bf*)(&Bs[nn][kseg])     = lo;
            *(v8bf*)(&Bs[nn][kseg + 8]) = hi;
        }
#endif
        // ---- stage row-major B tile (32 x 128) with register transpose
        if (!BK) {
            int kk   = tid >> 3;         // 0..31
            int nseg = (tid & 7) * 16;   // 0..112
            v8bf lo = zb, hi = zb;
            if (N0 + nseg < N) {  // N is a multiple of 16 in all uses
                const bf16* p = Bb + (long)(k0 + kk) * ldbr + (N0 + nseg);
                lo = *(const v8bf*)(p);
                hi = *(const v8bf*)(p + 8);
                if (k0 + BLK_K < Kd)  // prefetch next K tile of weights
                    __builtin_prefetch(p + (long)BLK_K * ldbr, 0, 1);
            }
#pragma unroll
            for (int i = 0; i < 8; ++i) Bs[nseg + i][kk] = lo[i];
#pragma unroll
            for (int i = 0; i < 8; ++i) Bs[nseg + 8 + i][kk] = hi[i];
        }
#if HAVE_TDM
        if (wave == 0) __builtin_amdgcn_s_wait_tensorcnt(0);
#endif
        __syncthreads();

        // ---- fragments + WMMA
        const int half = lane >> 4;
        const int l16  = lane & 15;
        union U { v16bf v; struct { v8bf lo, hi; } p; };
        U a[4], b[2];
#pragma unroll
        for (int mt = 0; mt < 4; ++mt) {
            const bf16* ap = &As[wm * 64 + mt * 16 + l16][0];
            a[mt].p.lo = *(const v8bf*)(ap + half * 8);
            a[mt].p.hi = *(const v8bf*)(ap + 16 + half * 8);
        }
#pragma unroll
        for (int nt = 0; nt < 2; ++nt) {
            const bf16* bp = &Bs[wn * 32 + nt * 16 + l16][0];
            b[nt].p.lo = *(const v8bf*)(bp + half * 8);
            b[nt].p.hi = *(const v8bf*)(bp + 16 + half * 8);
        }
#pragma unroll
        for (int mt = 0; mt < 4; ++mt)
#pragma unroll
            for (int nt = 0; nt < 2; ++nt)
                acc[mt][nt] = __builtin_amdgcn_wmma_f32_16x16x32_bf16(
                    false, a[mt].v, false, b[nt].v, (short)0, acc[mt][nt], false, false);
        __syncthreads();
    }

    // ---- epilogue
    const int half = lane >> 4;
    const int l16  = lane & 15;
    float* Cf = (float*)Cp + (long)bo * sCo + (long)bi * sCi;
    bf16*  Cb = (bf16*)Cp + (long)bo * sCo + (long)bi * sCi;
#pragma unroll
    for (int mt = 0; mt < 4; ++mt) {
#pragma unroll
        for (int nt = 0; nt < 2; ++nt) {
            int n = N0 + wn * 32 + nt * 16 + l16;
            if (n >= N) continue;
#pragma unroll
            for (int r = 0; r < 8; ++r) {
                int m = M0 + wm * 64 + mt * 16 + half * 8 + r;
                if (m >= M) continue;
                float v = acc[mt][nt][r];
                long idx = (long)m * ldc + n;
                if (EPI == EPI_F32)        Cf[idx] = v;
                else if (EPI == EPI_BF16)  Cb[idx] = f2b(v);
                else if (EPI == EPI_ADD)   Cf[idx] += v;
                else                       Cf[idx] += rs[(long)m * rs_stride] * v;
            }
        }
    }
}

// -------------------------------------------------- softmax over bf16 rows
__global__ void k_softmax_bf16(bf16* __restrict__ p, int rowlen, float scale) {
    long row = blockIdx.x;
    bf16* pr = p + row * (long)rowlen;
    __shared__ float red[8];
    int lane = threadIdx.x & 31, wid = threadIdx.x >> 5, nw = blockDim.x >> 5;

    float mx = -1e30f;
    for (int i = threadIdx.x; i < rowlen; i += blockDim.x)
        mx = fmaxf(mx, b2f(pr[i]) * scale);
    for (int off = 16; off > 0; off >>= 1) mx = fmaxf(mx, __shfl_down(mx, off, 32));
    if (lane == 0) red[wid] = mx;
    __syncthreads();
    if (threadIdx.x == 0) {
        float m = -1e30f;
        for (int i = 0; i < nw; ++i) m = fmaxf(m, red[i]);
        red[0] = m;
    }
    __syncthreads();
    mx = red[0];
    __syncthreads();

    float sum = 0.f;
    for (int i = threadIdx.x; i < rowlen; i += blockDim.x) {
        float v = __expf(b2f(pr[i]) * scale - mx);
        sum += v;
        pr[i] = f2b(v);
    }
    for (int off = 16; off > 0; off >>= 1) sum += __shfl_down(sum, off, 32);
    if (lane == 0) red[wid] = sum;
    __syncthreads();
    if (threadIdx.x == 0) {
        float s = 0.f;
        for (int i = 0; i < nw; ++i) s += red[i];
        red[0] = 1.f / s;
    }
    __syncthreads();
    float inv = red[0];
    for (int i = threadIdx.x; i < rowlen; i += blockDim.x)
        pr[i] = f2b(b2f(pr[i]) * inv);
}

// -------------------------------------------------------------- MoE router
// probs = softmax(h @ rW^T); top4; renormalized softmax of top-k probs -> cw[T,E]
__global__ void k_router(const bf16* __restrict__ xn, const float* __restrict__ rW,
                         float* __restrict__ cw) {
    int t = blockIdx.x;
    __shared__ float xs[D_];
    __shared__ float lg[E_];
    for (int d = threadIdx.x; d < D_; d += blockDim.x) xs[d] = b2f(xn[(long)t * D_ + d]);
    __syncthreads();
    // blockDim == 128: 16 groups of 8 threads, one expert each
    int grp = threadIdx.x >> 3;
    int gl  = threadIdx.x & 7;
    const float* wrow = rW + (long)grp * D_;
    float s = 0.f;
    for (int d = gl; d < D_; d += 8) s += xs[d] * wrow[d];
    for (int off = 4; off > 0; off >>= 1) s += __shfl_down(s, off, 8);
    if (gl == 0) lg[grp] = s;
    __syncthreads();
    if (threadIdx.x == 0) {
        float mx = -1e30f;
        for (int e = 0; e < E_; ++e) mx = fmaxf(mx, lg[e]);
        float p[E_], sum = 0.f;
        for (int e = 0; e < E_; ++e) { p[e] = __expf(lg[e] - mx); sum += p[e]; }
        for (int e = 0; e < E_; ++e) p[e] /= sum;
        int   idx[KTOP_];
        float w[KTOP_];
        bool  used[E_];
        for (int e = 0; e < E_; ++e) used[e] = false;
        for (int k = 0; k < KTOP_; ++k) {
            int best = 0; float bv = -1.f;
            for (int e = 0; e < E_; ++e)
                if (!used[e] && p[e] > bv) { bv = p[e]; best = e; }
            used[best] = true; idx[k] = best; w[k] = bv;
        }
        float m2 = w[0];  // sorted descending
        float s2 = 0.f;
        for (int k = 0; k < KTOP_; ++k) { w[k] = __expf(w[k] - m2); s2 += w[k]; }
        float out[E_];
        for (int e = 0; e < E_; ++e) out[e] = 0.f;
        for (int k = 0; k < KTOP_; ++k) out[idx[k]] = w[k] / s2;
        for (int e = 0; e < E_; ++e) cw[(long)t * E_ + e] = out[e];
    }
}

__global__ void k_silu_mul(const float* __restrict__ g, const float* __restrict__ u,
                           bf16* __restrict__ s, long n) {
    long i = (long)blockIdx.x * blockDim.x + threadIdx.x;
    long stride = (long)gridDim.x * blockDim.x;
    for (; i < n; i += stride) {
        float gv = g[i];
        float sv = gv / (1.f + __expf(-gv));
        s[i] = f2b(sv * u[i]);
    }
}

// pooled[b][d] = mean_s xf[b,s,d]
__global__ void k_mean(const float* __restrict__ xf, float* __restrict__ pooled) {
    int i = blockIdx.x * blockDim.x + threadIdx.x;
    if (i >= B_ * D_) return;
    int b = i / D_, d = i - b * D_;
    float s = 0.f;
    for (int t = 0; t < S_; ++t) s += xf[((long)b * S_ + t) * D_ + d];
    pooled[i] = s * (1.f / (float)S_);
}

__global__ void k_classify(const float* __restrict__ pooled, const float* __restrict__ cW,
                           const float* __restrict__ cb, float* __restrict__ out) {
    __shared__ float lg[B_][C_];
    int tid = threadIdx.x;
    if (tid < B_ * C_) {
        int b = tid / C_, c = tid - b * C_;
        float s = cb[c];
        for (int d = 0; d < D_; ++d) s += pooled[b * D_ + d] * cW[(long)c * D_ + d];
        lg[b][c] = s;
    }
    __syncthreads();
    if (tid < B_) {
        float mx = -1e30f;
        for (int c = 0; c < C_; ++c) mx = fmaxf(mx, lg[tid][c]);
        float e[C_], s = 0.f;
        for (int c = 0; c < C_; ++c) { e[c] = __expf(lg[tid][c] - mx); s += e[c]; }
        for (int c = 0; c < C_; ++c) out[tid * C_ + c] = e[c] / s;
    }
}

// ------------------------------------------------------------------- host
template <int EPI, bool BK>
static void launch_gemm(hipStream_t st, const bf16* A, long lda, long sAo, long sAi,
                        const bf16* Bm, long ldbr, long ldbc, long sBo, long sBi,
                        void* C, long ldc, long sCo, long sCi,
                        const float* rs, int rss,
                        int M, int N, int Kd, int batches, int nInner) {
    dim3 grid((N + BLK_N - 1) / BLK_N, (M + BLK_M - 1) / BLK_M, batches);
    k_gemm<EPI, BK><<<grid, 256, 0, st>>>(A, lda, sAo, sAi, Bm, ldbr, ldbc, sBo, sBi,
                                          C, ldc, sCo, sCi, rs, rss, M, N, Kd, nInner);
}

extern "C" void kernel_launch(void* const* d_in, const int* in_sizes, int n_in,
                              void* d_out, int out_size, void* d_ws, size_t ws_size,
                              hipStream_t stream) {
    const int*   X    = (const int*)d_in[0];
    const float* emb  = (const float*)d_in[1];
    const float* Wq   = (const float*)d_in[2];
    const float* Wk   = (const float*)d_in[3];
    const float* Wv   = (const float*)d_in[4];
    const float* Wo   = (const float*)d_in[5];
    const float* n1w  = (const float*)d_in[6];
    const float* n2w  = (const float*)d_in[7];
    const float* rW   = (const float*)d_in[8];
    const float* sW1  = (const float*)d_in[9];
    const float* sW3  = (const float*)d_in[10];
    const float* sW2  = (const float*)d_in[11];
    const float* eW1  = (const float*)d_in[12];
    const float* eW3  = (const float*)d_in[13];
    const float* eW2  = (const float*)d_in[14];
    const float* fnw  = (const float*)d_in[15];
    const float* cW   = (const float*)d_in[16];
    const float* cb   = (const float*)d_in[17];

    // bump allocator over workspace (256B aligned)
    char* wsb = (char*)d_ws;
    size_t off = 0;
    auto alloc = [&](size_t bytes) -> void* {
        void* p = wsb + off;
        off = (off + bytes + 255) & ~(size_t)255;
        return p;
    };

    const long NWQ = (long)L_ * D_ * D_;
    const long NSW = (long)L_ * D_ * DFF_;
    const long NEW = (long)L_ * E_ * D_ * DFF_;

    bf16* wq_b  = (bf16*)alloc(NWQ * 2);
    bf16* wk_b  = (bf16*)alloc(NWQ * 2);
    bf16* wv_b  = (bf16*)alloc(NWQ * 2);
    bf16* wo_b  = (bf16*)alloc(NWQ * 2);
    bf16* sw1_b = (bf16*)alloc(NSW * 2);
    bf16* sw3_b = (bf16*)alloc(NSW * 2);
    bf16* sw2_b = (bf16*)alloc(NSW * 2);
    bf16* ew1_b = (bf16*)alloc(NEW * 2);
    bf16* ew3_b = (bf16*)alloc(NEW * 2);
    bf16* ew2_b = (bf16*)alloc(NEW * 2);

    float* x    = (float*)alloc((long)T_ * D_ * 4);
    bf16*  xn   = (bf16*)alloc((long)T_ * D_ * 2);
    bf16*  qb   = (bf16*)alloc((long)T_ * D_ * 2);
    bf16*  kb   = (bf16*)alloc((long)T_ * D_ * 2);
    bf16*  vb   = (bf16*)alloc((long)T_ * D_ * 2);
    bf16*  ob   = (bf16*)alloc((long)T_ * D_ * 2);
    bf16*  sc   = (bf16*)alloc((long)B_ * NH_ * S_ * S_ * 2);
    float* gbuf = (float*)alloc((long)T_ * DFF_ * 4);
    float* ubuf = (float*)alloc((long)T_ * DFF_ * 4);
    bf16*  sbuf = (bf16*)alloc((long)T_ * DFF_ * 2);
    float* cw   = (float*)alloc((long)T_ * E_ * 4);
    float* xf   = (float*)alloc((long)T_ * D_ * 4);
    float* pool = (float*)alloc((long)B_ * D_ * 4);

    auto conv = [&](const float* src, bf16* dst, long n) {
        long blocks = (n + 256 * 8 - 1) / (256 * 8);
        if (blocks > 16384) blocks = 16384;
        if (blocks < 1) blocks = 1;
        k_f32_to_bf16<<<dim3((unsigned)blocks), 256, 0, stream>>>(src, dst, n);
    };
    conv(Wq, wq_b, NWQ);   conv(Wk, wk_b, NWQ);
    conv(Wv, wv_b, NWQ);   conv(Wo, wo_b, NWQ);
    conv(sW1, sw1_b, NSW); conv(sW3, sw3_b, NSW); conv(sW2, sw2_b, NSW);
    conv(eW1, ew1_b, NEW); conv(eW3, ew3_b, NEW); conv(eW2, ew2_b, NEW);

    k_embed<<<T_, 256, 0, stream>>>(X, emb, x);

    const long SD  = (long)S_ * D_;
    const long SS  = (long)S_ * S_;
    const long DD  = (long)D_ * D_;
    const long DF  = (long)D_ * DFF_;

    for (int l = 0; l < L_; ++l) {
        // ---- attention
        k_rmsnorm<<<T_, 256, 0, stream>>>(x, n1w + (long)l * D_, xn, nullptr);
        launch_gemm<EPI_BF16, false>(stream, xn, D_, 0, 0, wq_b + l * DD, D_, 1, 0, 0,
                                     qb, D_, 0, 0, nullptr, 0, T_, D_, D_, 1, 1);
        launch_gemm<EPI_BF16, false>(stream, xn, D_, 0, 0, wk_b + l * DD, D_, 1, 0, 0,
                                     kb, D_, 0, 0, nullptr, 0, T_, D_, D_, 1, 1);
        launch_gemm<EPI_BF16, false>(stream, xn, D_, 0, 0, wv_b + l * DD, D_, 1, 0, 0,
                                     vb, D_, 0, 0, nullptr, 0, T_, D_, D_, 1, 1);
        // scores[b,h] = q_bh (S x dk) @ k_bh^T (dk x S)
        launch_gemm<EPI_BF16, true>(stream, qb, D_, SD, DK_, kb, 1, D_, SD, DK_,
                                    sc, S_, (long)NH_ * SS, SS, nullptr, 0,
                                    S_, S_, DK_, B_ * NH_, NH_);
        k_softmax_bf16<<<B_ * NH_ * S_, 256, 0, stream>>>(sc, S_, 0.125f);
        // o_bh = probs (S x S) @ v_bh (S x dk)
        launch_gemm<EPI_BF16, false>(stream, sc, S_, (long)NH_ * SS, SS, vb, D_, 1, SD, DK_,
                                     ob, D_, SD, DK_, nullptr, 0,
                                     S_, DK_, S_, B_ * NH_, NH_);
        // x += o @ Wo
        launch_gemm<EPI_ADD, false>(stream, ob, D_, 0, 0, wo_b + l * DD, D_, 1, 0, 0,
                                    x, D_, 0, 0, nullptr, 0, T_, D_, D_, 1, 1);

        // ---- MoE
        k_rmsnorm<<<T_, 256, 0, stream>>>(x, n2w + (long)l * D_, xn, nullptr);
        k_router<<<T_, 128, 0, stream>>>(xn, rW + (long)l * E_ * D_, cw);

        // shared expert
        launch_gemm<EPI_F32, false>(stream, xn, D_, 0, 0, sw1_b + l * DF, DFF_, 1, 0, 0,
                                    gbuf, DFF_, 0, 0, nullptr, 0, T_, DFF_, D_, 1, 1);
        launch_gemm<EPI_F32, false>(stream, xn, D_, 0, 0, sw3_b + l * DF, DFF_, 1, 0, 0,
                                    ubuf, DFF_, 0, 0, nullptr, 0, T_, DFF_, D_, 1, 1);
        k_silu_mul<<<4096, 256, 0, stream>>>(gbuf, ubuf, sbuf, (long)T_ * DFF_);
        launch_gemm<EPI_ADD, false>(stream, sbuf, DFF_, 0, 0, sw2_b + l * DF, D_, 1, 0, 0,
                                    x, D_, 0, 0, nullptr, 0, T_, D_, DFF_, 1, 1);

        // routed experts (dense over E, weighted by combine weights -> exact)
        for (int e = 0; e < E_; ++e) {
            const bf16* e1 = ew1_b + ((long)l * E_ + e) * DF;
            const bf16* e3 = ew3_b + ((long)l * E_ + e) * DF;
            const bf16* e2 = ew2_b + ((long)l * E_ + e) * DF;
            launch_gemm<EPI_F32, false>(stream, xn, D_, 0, 0, e1, DFF_, 1, 0, 0,
                                        gbuf, DFF_, 0, 0, nullptr, 0, T_, DFF_, D_, 1, 1);
            launch_gemm<EPI_F32, false>(stream, xn, D_, 0, 0, e3, DFF_, 1, 0, 0,
                                        ubuf, DFF_, 0, 0, nullptr, 0, T_, DFF_, D_, 1, 1);
            k_silu_mul<<<4096, 256, 0, stream>>>(gbuf, ubuf, sbuf, (long)T_ * DFF_);
            launch_gemm<EPI_ADD_SCALED, false>(stream, sbuf, DFF_, 0, 0, e2, D_, 1, 0, 0,
                                               x, D_, 0, 0, cw + e, E_,
                                               T_, D_, DFF_, 1, 1);
        }
    }

    // ---- head
    k_rmsnorm<<<T_, 256, 0, stream>>>(x, fnw, nullptr, xf);
    k_mean<<<(B_ * D_ + 255) / 256, 256, 0, stream>>>(xf, pool);
    k_classify<<<1, 64, 0, stream>>>(pool, cW, cb, (float*)d_out);

    (void)in_sizes; (void)n_in; (void)out_size; (void)ws_size;
}
